// CIN_16612933501310
// MI455X (gfx1250) — matrix operand order repaired and checked by reading, not compiled, not run on hardware
//
#include <hip/hip_runtime.h>
#include <hip/hip_bf16.h>

// ---------------------------------------------------------------------------
// CIN (cross-interaction network) fused for gfx1250 (MI455X), wave32 + WMMA
// + async global->LDS DMA for the weight tiles.
//
//   Xn[b,k,d] = relu( sum_{m,n} Xk[b,m,d] * x[b,n,d] * W[m,n,k] )   (2 layers)
//   y[b,p,d]  = x[b,p,d] + proj_b[p] + sum_t proj_w[p,t]*cin[t,d]
//   out       = LayerNorm_d(y)
//
// GEMM view per layer: rows = (b,d) [16384], K = m*64+n [4096], cols = k [64].
// A[(b,d), m*64+n] = Xk[b,m,d]*x[b,n,d] is formed on the fly with v_pk_mul_f16.
// ---------------------------------------------------------------------------

typedef _Float16 v16h __attribute__((ext_vector_type(16)));
typedef _Float16 v8h  __attribute__((ext_vector_type(8)));
typedef _Float16 v2h  __attribute__((ext_vector_type(2)));
typedef float    v8f  __attribute__((ext_vector_type(8)));

union H16 { v16h v; v8h h[2]; v2h p[8]; };

// GCC-style int4 vector with explicit address spaces (matches builtin sig
// "vV4i*1V4i*3IiIi": (global int4*, lds int4*, imm offset, imm cpol))
typedef int v4i __attribute__((vector_size(16)));
typedef __attribute__((address_space(1))) v4i as1_v4i;
typedef __attribute__((address_space(3))) v4i as3_v4i;

#define UP 88          // LDS row pitch (halves) for x / Xn tiles: 176B, 16B-aligned,
                       // 44-dword stride -> conflict-free for 16-row b128 reads
#define BP 40          // LDS pitch (halves) per (chunk,k) row of W: 80B, 16B-aligned,
                       // 20-dword stride -> conflict-free for 16-lane b128 reads
#define CH 4           // chunks staged per phase
#define BUF_F4 1280    // float4 per W stage buffer (CH*64*5)

// --- CDNA5 async global->LDS (ASYNCcnt) -------------------------------------
__device__ __forceinline__ void async_ld16(void* lds, const void* g) {
#if __has_builtin(__builtin_amdgcn_global_load_async_to_lds_b128)
  __builtin_amdgcn_global_load_async_to_lds_b128(
      (as1_v4i*)(unsigned long long)g,
      (as3_v4i*)(unsigned)(unsigned long long)lds, 0, 0);
#else
  unsigned l = (unsigned)(unsigned long long)lds;
  asm volatile("global_load_async_to_lds_b128 %0, %1, off"
               :: "v"(l), "v"(g) : "memory");
#endif
}

#if __has_builtin(__builtin_amdgcn_s_wait_asynccnt)
#define WAIT_ASYNC(n) __builtin_amdgcn_s_wait_asynccnt(n)
#else
#define WAIT_ASYNC(n) asm volatile("s_wait_asynccnt %0" :: "n"(n) : "memory")
#endif

// --- prep: W[m,n,k] f32  ->  Wt[chunk][k][K] f16, chunk=m*2+(n>=32), K=n%32 ---
__global__ __launch_bounds__(256) void cin_prep(
    const float* __restrict__ W1, const float* __restrict__ W2,
    const float* __restrict__ Pw,
    _Float16* __restrict__ Wt1, _Float16* __restrict__ Wt2,
    _Float16* __restrict__ Pt)
{
  int e = blockIdx.x * 256 + threadIdx.x;
  if (e < 2 * 262144) {
    const float* W = (e < 262144) ? W1 : W2;
    _Float16*   Wt = (e < 262144) ? Wt1 : Wt2;
    int ee = e & (262144 - 1);
    int c  = ee >> 11;          // chunk 0..127
    int k  = (ee >> 5) & 63;    // output col
    int K  = ee & 31;           // pos within chunk
    int m  = c >> 1;
    int n  = (c & 1) * 32 + K;
    Wt[ee] = (_Float16)W[(m * 64 + n) * 64 + k];
  } else if (e < 2 * 262144 + 8192) {
    int ee = e - 2 * 262144;
    Pt[ee] = (_Float16)Pw[ee];  // proj_w already [p][t]
  }
}

// issue one phase (CH chunks = CH*64*32 halves) of W as async DMA into LDS
__device__ __forceinline__ void stage_async(const float4* gw, int phase, int bufSel,
                                            float4* Bs4, int w, int lane)
{
#pragma unroll
  for (int i = 0; i < 4; ++i) {
    int g  = (i * 8 + w) * 32 + lane;           // 0..1023 float4 groups
    int K8 = g & 3, kk = (g >> 2) & 63, cc = g >> 8;
    async_ld16(Bs4 + bufSel * BUF_F4 + (cc * 64 + kk) * 5 + K8,
               gw + phase * (CH * 256) + g);
  }
}

// --- one cross layer: 128 K-chunks, double-buffered async staging -----------
__device__ __forceinline__ void cin_layer(
    const _Float16* __restrict__ Wt,   // global Wt table (f16)
    const _Float16* xsr,               // LDS: x rows   xsr[r*UP + n]
    const _Float16* usrc,              // LDS: Xk rows  usrc[r*UP + m]
    _Float16* dstS,                    // LDS: relu(out) -> dstS[r*UP + k]
    float4* Bs4, const _Float16* Bsh,
    int tid, int r, int l15, int hi, int kh)
{
  int w = tid >> 5, lane = tid & 31;
  const float4* gw = (const float4*)Wt;
  v8f c0 = {}; v8f c1 = {};

  stage_async(gw, 0, 0, Bs4, w, lane);
  for (int ph = 0; ph < 32; ++ph) {
    int cur = ph & 1;
    if (ph < 31) {
      stage_async(gw, ph + 1, cur ^ 1, Bs4, w, lane);
      WAIT_ASYNC(4);                   // phase ph's 4 DMAs (in-order) done
    } else {
      WAIT_ASYNC(0);
    }
    __syncthreads();                   // all waves' DMA visible
    const _Float16* Bbase = Bsh + cur * (BUF_F4 * 8);
#pragma unroll
    for (int cc = 0; cc < CH; ++cc) {
      int chunk = ph * CH + cc;
      int m  = chunk >> 1;
      int n0 = (chunk & 1) * 32;
      // A fragment: lane (l15,hi): M=l15-row, K = hi*16 + 0..15 contiguous
      _Float16 u = usrc[r * UP + m];
      v2h uu; uu[0] = u; uu[1] = u;
      v8h vlo = *(const v8h*)&xsr[r * UP + n0 + hi * 16];
      v8h vhi = *(const v8h*)&xsr[r * UP + n0 + hi * 16 + 8];
      const v2h* lp = (const v2h*)&vlo;
      const v2h* hp = (const v2h*)&vhi;
      H16 A;
#pragma unroll
      for (int j = 0; j < 4; ++j) { A.p[j] = lp[j] * uu; A.p[4 + j] = hp[j] * uu; }
      // B fragments: lane: N-col = l15, K = hi*16 + 0..15 contiguous
#pragma unroll
      for (int kt = 0; kt < 2; ++kt) {
        int kcol = (kh * 2 + kt) * 16 + l15;
        const _Float16* bp = Bbase + (cc * 64 + kcol) * BP + hi * 16;
        H16 Bf; Bf.h[0] = *(const v8h*)bp; Bf.h[1] = *(const v8h*)(bp + 8);
        if (kt == 0)
          c0 = __builtin_amdgcn_wmma_f32_16x16x32_f16(false, A.v, false, Bf.v,
                                                      (short)0, c0, false, false);
        else
          c1 = __builtin_amdgcn_wmma_f32_16x16x32_f16(false, A.v, false, Bf.v,
                                                      (short)0, c1, false, false);
      }
    }
    __syncthreads();                   // readers done before buffer reuse
  }
  // relu + scatter to LDS as f16 (C layout: VGPR j -> row hi*8+j, col = l15)
  int rbase = r - l15;  // rt*16
#pragma unroll
  for (int j = 0; j < 8; ++j) {
    int row = rbase + hi * 8 + j;
    dstS[row * UP + (kh * 2 + 0) * 16 + l15] = (_Float16)fmaxf(c0[j], 0.0f);
    dstS[row * UP + (kh * 2 + 1) * 16 + l15] = (_Float16)fmaxf(c1[j], 0.0f);
  }
}

// --- main fused kernel: 256 WGs x 256 threads (8 waves) ---------------------
__global__ __launch_bounds__(256) void cin_main(
    const float* __restrict__ x,
    const _Float16* __restrict__ Wt1,
    const _Float16* __restrict__ Wt2,
    const _Float16* __restrict__ Pt,
    const float* __restrict__ pb,
    float* __restrict__ yout)
{
  __shared__ __align__(16) _Float16 xsr[64 * UP];
  __shared__ __align__(16) _Float16 u1s[64 * UP];
  __shared__ __align__(16) float4  BsBuf[2 * BUF_F4];    // 2 x 20 KB W stage
  _Float16* Bsh = (_Float16*)BsBuf;
  _Float16* u2s = (_Float16*)BsBuf;                      // reuse after layer 2

  int tid = threadIdx.x;
  int wg  = blockIdx.x;
  int b   = wg >> 3;
  int d0  = (wg & 7) * 64;
  int w   = tid >> 5;
  int rt  = w >> 1;              // row tile 0..3 (16 rows each)
  int kh  = w & 1;               // k half (32 cols)
  int l15 = tid & 15;
  int hi  = (tid >> 4) & 1;
  int r   = rt * 16 + l15;

  // stage x[b, n, d0..d0+63] -> xsr[r][n] (f16)
#pragma unroll
  for (int i = 0; i < 16; ++i) {
    int e = i * 256 + tid;
    int rr = e & 63, n = e >> 6;
    xsr[rr * UP + n] = (_Float16)x[(b * 64 + n) * 512 + d0 + rr];
  }
  __syncthreads();

  cin_layer(Wt1, xsr, xsr, u1s, BsBuf, Bsh, tid, r, l15, hi, kh);   // Xn1
  __syncthreads();
  cin_layer(Wt2, xsr, u1s, u2s, BsBuf, Bsh, tid, r, l15, hi, kh);   // Xn2
  __syncthreads();

  // projection: y = x + b + proj_w[p,0:64]·Xn1 + proj_w[p,64:128]·Xn2
  for (int tt = w; tt < 16; tt += 8) {
    int pt  = tt >> 2;           // p tile 0..3
    int rt2 = tt & 3;            // row tile 0..3
    int rr  = rt2 * 16 + l15;
    v8f c = {};
#pragma unroll
    for (int j = 0; j < 8; ++j) {
      int p = pt * 16 + hi * 8 + j;
      c[j] = x[(b * 64 + p) * 512 + d0 + rr] + pb[p];
    }
#pragma unroll
    for (int c4 = 0; c4 < 4; ++c4) {
      const _Float16* pa = Pt + (pt * 16 + l15) * 128 + c4 * 32 + hi * 16;
      H16 A; A.h[0] = *(const v8h*)pa; A.h[1] = *(const v8h*)(pa + 8);
      const _Float16* ubase = (c4 < 2) ? u1s : u2s;
      int t0 = (c4 & 1) * 32 + hi * 16;
      const _Float16* bp = ubase + rr * UP + t0;
      H16 Bf; Bf.h[0] = *(const v8h*)bp; Bf.h[1] = *(const v8h*)(bp + 8);
      c = __builtin_amdgcn_wmma_f32_16x16x32_f16(false, A.v, false, Bf.v,
                                                 (short)0, c, false, false);
    }
#pragma unroll
    for (int j = 0; j < 8; ++j) {
      int p = pt * 16 + hi * 8 + j;
      yout[(b * 64 + p) * 512 + d0 + rr] = c[j];
    }
  }
}

// --- LayerNorm over d (in-place on d_out), one 512-row per block ------------
__global__ __launch_bounds__(128) void cin_ln(
    float* __restrict__ y, const float* __restrict__ g, const float* __restrict__ be)
{
  int row = blockIdx.x;
  int tid = threadIdx.x;
  float4* rp = (float4*)(y + row * 512);
  float4 v = rp[tid];
  float s  = v.x + v.y + v.z + v.w;
  float s2 = v.x * v.x + v.y * v.y + v.z * v.z + v.w * v.w;
#pragma unroll
  for (int off = 16; off > 0; off >>= 1) {
    s  += __shfl_down(s, off);
    s2 += __shfl_down(s2, off);
  }
  __shared__ float sa[4], sb[4];
  if ((tid & 31) == 0) { sa[tid >> 5] = s; sb[tid >> 5] = s2; }
  __syncthreads();
  float ts = sa[0] + sa[1] + sa[2] + sa[3];
  float t2 = sb[0] + sb[1] + sb[2] + sb[3];
  float mu  = ts * (1.0f / 512.0f);
  float var = t2 * (1.0f / 512.0f) - mu * mu;
  float inv = rsqrtf(var + 1e-5f);
  int c0 = tid * 4;
  v.x = g[c0 + 0] * (v.x - mu) * inv + be[c0 + 0];
  v.y = g[c0 + 1] * (v.y - mu) * inv + be[c0 + 1];
  v.z = g[c0 + 2] * (v.z - mu) * inv + be[c0 + 2];
  v.w = g[c0 + 3] * (v.w - mu) * inv + be[c0 + 3];
  rp[tid] = v;
}

extern "C" void kernel_launch(void* const* d_in, const int* in_sizes, int n_in,
                              void* d_out, int out_size, void* d_ws, size_t ws_size,
                              hipStream_t stream)
{
  const float* x  = (const float*)d_in[0];
  const float* W1 = (const float*)d_in[1];
  const float* W2 = (const float*)d_in[2];
  const float* Pw = (const float*)d_in[3];
  const float* pb = (const float*)d_in[4];
  const float* g  = (const float*)d_in[5];
  const float* be = (const float*)d_in[6];
  float* out = (float*)d_out;

  _Float16* Wt1 = (_Float16*)d_ws;          // 512 KB
  _Float16* Wt2 = Wt1 + 262144;             // 512 KB
  _Float16* Pt  = Wt1 + 524288;             // 16 KB

  cin_prep<<<2080, 256, 0, stream>>>(W1, W2, Pw, Wt1, Wt2, Pt);
  cin_main<<<256, 256, 0, stream>>>(x, Wt1, Wt2, Pt, pb, out);
  cin_ln<<<2048, 128, 0, stream>>>(out, g, be);
}
